// Net_5634997093329
// MI455X (gfx1250) — matrix operand order repaired and checked by reading, not compiled, not run on hardware
//
#include <hip/hip_runtime.h>
#include <hip/hip_bf16.h>

// ---------------- problem constants (from the reference) ----------------
#define HD   30                         // hidden channels
#define BG   64                         // graphs
#define NPG  2048                       // nodes per graph
#define DEG  16
#define NTOT (BG * NPG)                 // 131072 nodes
#define ETOT ((long long)NTOT * DEG)    // 2,097,152 edges
#define KP1  1844
#define KP2  1660
#define KP3  1494
#define NHD  11
#define ODIM 8

typedef __attribute__((ext_vector_type(16))) __bf16 v16bf;
typedef __attribute__((ext_vector_type(8)))  __bf16 v8bf;
typedef __attribute__((ext_vector_type(8)))  float  v8f;

static __device__ __forceinline__ v16bf frag2(v8bf lo, v8bf hi) {
  return __builtin_shufflevector(lo, hi, 0, 1, 2, 3, 4, 5, 6, 7, 8, 9, 10, 11, 12, 13, 14, 15);
}

// ---------------------------- elementwise / prep kernels ----------------------------

__global__ void fill_kernel(float* p, float v, long long n) {
  long long i = (long long)blockIdx.x * blockDim.x + threadIdx.x;
  if (i < n) p[i] = v;
}

__global__ void relu_kernel(float* p, long long n) {
  long long i = (long long)blockIdx.x * blockDim.x + threadIdx.x;
  if (i < n) p[i] = fmaxf(p[i], 0.0f);
}

// fp32 [rows,K] -> bf16 [rows,Kp], zero padded (Kp multiple of 32 -> 16B-aligned rows)
__global__ void cvt_pad_kernel(const float* __restrict__ src, __bf16* __restrict__ dst,
                               long long rows, int K, int Kp) {
  long long idx = (long long)blockIdx.x * blockDim.x + threadIdx.x;
  if (idx >= rows * (long long)Kp) return;
  long long r = idx / Kp;
  int c = (int)(idx - r * Kp);
  dst[idx] = (c < K) ? (__bf16)src[r * (long long)K + c] : (__bf16)0.0f;
}

// pack fp32 W[K,N] (row-major) into WMMA B-fragment layout:
// P[(kt*NT + nt)*32 + lane][16] ; K=kt*32+8*(lane>>4)+(j&7)+16*(j>>3) ; col=nt*16+(lane&15)
__global__ __launch_bounds__(32)
void pack_w_kernel(const float* __restrict__ W, long long wstride,
                   __bf16* __restrict__ P, long long pstride,
                   int K, int N, int NT) {
  int lane = threadIdx.x;
  const float* Wh = W + wstride * blockIdx.z;
  __bf16* Ph = P + pstride * blockIdx.z;
  int col = blockIdx.y * 16 + (lane & 15);
  int khi = (lane >> 4) << 3;
  __bf16* out = Ph + (((size_t)blockIdx.x * NT + blockIdx.y) * 32 + lane) * 16;
#pragma unroll
  for (int j = 0; j < 16; ++j) {
    int k = blockIdx.x * 32 + (j & 7) + ((j >> 3) << 4) + khi;
    float v = (k < K && col < N) ? Wh[(size_t)k * N + col] : 0.0f;
    out[j] = (__bf16)v;
  }
}

// one wave per node; IN=128 -> 4 feats/lane
__global__ void layernorm_kernel(const float* __restrict__ x, const float* __restrict__ g,
                                 const float* __restrict__ b, float* __restrict__ out, int N) {
  int node = (int)((blockIdx.x * (long long)blockDim.x + threadIdx.x) >> 5);
  int lane = threadIdx.x & 31;
  if (node >= N) return;
  const float* xr = x + (size_t)node * 128;
  float v[4]; float s = 0.f;
#pragma unroll
  for (int i = 0; i < 4; ++i) { v[i] = xr[lane + 32 * i]; s += v[i]; }
#pragma unroll
  for (int m = 16; m > 0; m >>= 1) s += __shfl_xor(s, m, 32);
  float mean = s * (1.0f / 128.0f);
  float vs = 0.f;
#pragma unroll
  for (int i = 0; i < 4; ++i) { float d = v[i] - mean; vs += d * d; }
#pragma unroll
  for (int m = 16; m > 0; m >>= 1) vs += __shfl_xor(vs, m, 32);
  float rstd = rsqrtf(vs * (1.0f / 128.0f) + 1e-5f);
  float* o = out + (size_t)node * 128;
#pragma unroll
  for (int i = 0; i < 4; ++i) {
    int f = lane + 32 * i;
    o[f] = (v[i] - mean) * rstd * g[f] + b[f];
  }
}

// edge scatter: agg[dst] += x[src] * ew   (one thread per (edge,feature))
__global__ void scatter_kernel(const float* __restrict__ xin, const int* __restrict__ src,
                               const int* __restrict__ dst, const float* __restrict__ ew,
                               float* __restrict__ agg, long long E, int F) {
  long long idx = (long long)blockIdx.x * blockDim.x + threadIdx.x;
  if (idx >= E * (long long)F) return;
  long long e = idx / F;
  int f = (int)(idx - e * F);
  float w = ew[e];
  if (w == 0.0f) return;                         // dropped edges contribute nothing
  __builtin_prefetch(src + e + 1024, 0, 0);      // global_prefetch ahead in edge list
  float v = xin[(size_t)src[e] * F + f] * w;
  atomicAdd(&agg[(size_t)dst[e] * F + f], v);
}

// ---------------------------- WMMA GEMM kernels ----------------------------
// out[:,0:30] = relu(agg @ Wr + xin @ Ws + br). bf16 inputs padded to Kp; packed weights (NT=2).
// One wave computes a 16x32 output tile: A fragments loaded once, 4 WMMAs per K-step.
__global__ __launch_bounds__(32)
void gconv_mm_kernel(const __bf16* __restrict__ aggb, const __bf16* __restrict__ xinb, int Kp,
                     const __bf16* __restrict__ Pr, const __bf16* __restrict__ Ps,
                     const float* __restrict__ br, float* __restrict__ out, int H) {
  const int NT = 2;
  int lane = threadIdx.x;
  int row = blockIdx.x * 16 + (lane & 15);
  int sel = (lane >> 4) << 3;                    // upper half-wave holds K+8 / K+24 runs
  const __bf16* arow = aggb + (size_t)row * Kp + sel;
  const __bf16* xrow = xinb + (size_t)row * Kp + sel;
  v8f acc0 = {}, acc1 = {};
  int KT = Kp >> 5;
  for (int kt = 0; kt < KT; ++kt) {
    const __bf16* a0 = arow + kt * 32;
    const __bf16* x0 = xrow + kt * 32;
    v16bf aA = frag2(*(const v8bf*)a0, *(const v8bf*)(a0 + 16));
    v16bf aX = frag2(*(const v8bf*)x0, *(const v8bf*)(x0 + 16));
    const __bf16* pr = Pr + ((size_t)kt * NT * 32 + lane) * 16;
    const __bf16* ps = Ps + ((size_t)kt * NT * 32 + lane) * 16;
    v16bf bR0 = frag2(*(const v8bf*)pr, *(const v8bf*)(pr + 8));
    v16bf bR1 = frag2(*(const v8bf*)(pr + 512), *(const v8bf*)(pr + 520));
    v16bf bS0 = frag2(*(const v8bf*)ps, *(const v8bf*)(ps + 8));
    v16bf bS1 = frag2(*(const v8bf*)(ps + 512), *(const v8bf*)(ps + 520));
    acc0 = __builtin_amdgcn_wmma_f32_16x16x32_bf16(false, aA, false, bR0, (short)0, acc0, false, false);
    acc1 = __builtin_amdgcn_wmma_f32_16x16x32_bf16(false, aA, false, bR1, (short)0, acc1, false, false);
    acc0 = __builtin_amdgcn_wmma_f32_16x16x32_bf16(false, aX, false, bS0, (short)0, acc0, false, false);
    acc1 = __builtin_amdgcn_wmma_f32_16x16x32_bf16(false, aX, false, bS1, (short)0, acc1, false, false);
  }
  int c0 = lane & 15;
  int rb = (lane >> 4) << 3;
  long long row0 = (long long)blockIdx.x * 16 + rb;
  float b0 = br[c0];
#pragma unroll
  for (int v = 0; v < 8; ++v)
    out[(row0 + v) * H + c0] = fmaxf(acc0[v] + b0, 0.0f);
  int c1 = 16 + c0;
  if (c1 < H) {
    float b1 = br[c1];
#pragma unroll
    for (int v = 0; v < 8; ++v)
      out[(row0 + v) * H + c1] = fmaxf(acc1[v] + b1, 0.0f);
  }
}

// per-head GEMM: C[h] = act(A[h] @ W[h] + bias[h]); A bf16 padded [*,Kp], W packed fragments.
__global__ __launch_bounds__(32)
void head_gemm_kernel(const __bf16* __restrict__ A, long long strideA, int Kp,
                      const __bf16* __restrict__ P, long long strideP, int NT,
                      const float* __restrict__ bias, long long strideB,
                      float* __restrict__ C, long long strideC,
                      int N, int doRelu) {
  int h = blockIdx.z;
  const __bf16* Ah = A + strideA * h;            // strideA==0 => shared A
  const __bf16* Ph = P + strideP * h;
  const float* bh = bias + strideB * h;
  float* Ch = C + strideC * h;
  int lane = threadIdx.x;
  int row = blockIdx.x * 16 + (lane & 15);
  int sel = (lane >> 4) << 3;
  const __bf16* arow = Ah + (size_t)row * Kp + sel;
  v8f acc = {};
  int KT = Kp >> 5;
  for (int kt = 0; kt < KT; ++kt) {
    const __bf16* a0 = arow + kt * 32;
    v16bf aF = frag2(*(const v8bf*)a0, *(const v8bf*)(a0 + 16));
    const __bf16* pb = Ph + (((size_t)kt * NT + blockIdx.y) * 32 + lane) * 16;
    v16bf bF = frag2(*(const v8bf*)pb, *(const v8bf*)(pb + 8));
    acc = __builtin_amdgcn_wmma_f32_16x16x32_bf16(false, aF, false, bF, (short)0, acc, false, false);
  }
  int col = blockIdx.y * 16 + (lane & 15);
  if (col < N) {
    float b = bh[col];
    int rb = (lane >> 4) << 3;
    long long row0 = (long long)blockIdx.x * 16 + rb;
#pragma unroll
    for (int v = 0; v < 8; ++v) {
      float r = acc[v] + b;
      Ch[row0 * N + (long long)v * N + col] = doRelu ? fmaxf(r, 0.0f) : r;
    }
  }
}

// ---------------------------- pooling / readout ----------------------------

// score = tanh(x.w / ||w||); one wave per node (H=30 <= 32 lanes)
__global__ void score_kernel(const float* __restrict__ xin, const float* __restrict__ w,
                             float* __restrict__ score, int N, int H) {
  int node = (int)((blockIdx.x * (long long)blockDim.x + threadIdx.x) >> 5);
  int lane = threadIdx.x & 31;
  if (node >= N) return;
  float wv = (lane < H) ? w[lane] : 0.0f;
  float xv = (lane < H) ? xin[(size_t)node * H + lane] : 0.0f;
  float d = xv * wv, n2 = wv * wv;
#pragma unroll
  for (int m = 16; m > 0; m >>= 1) { d += __shfl_xor(d, m, 32); n2 += __shfl_xor(n2, m, 32); }
  if (lane == 0) score[node] = tanhf(d * rsqrtf(n2));
}

// per-graph top-k via bitonic sort of 2048 (score,idx) pairs in LDS (16 KB of 320 KB/WGP)
__global__ __launch_bounds__(1024)
void topk_kernel(const float* __restrict__ score, int n, int k,
                 int* __restrict__ newidx, int* __restrict__ perm, float* __restrict__ gate) {
  __shared__ float ssc[2048];
  __shared__ int   sid[2048];
  int g = blockIdx.x;
  const float* sg = score + (size_t)g * n;
  for (int i = threadIdx.x; i < 2048; i += blockDim.x) {
    bool ok = i < n;
    ssc[i] = ok ? sg[i] : -3.4e38f;
    sid[i] = ok ? i : 0;
    if (ok) newidx[(size_t)g * n + i] = -1;
  }
  __syncthreads();
  for (int size = 2; size <= 2048; size <<= 1) {
    for (int stride = size >> 1; stride > 0; stride >>= 1) {
      for (int i = threadIdx.x; i < 2048; i += blockDim.x) {
        int p = i ^ stride;
        if (p > i) {
          bool desc = ((i & size) == 0);
          float a = ssc[i], c = ssc[p];
          if (desc ? (a < c) : (a > c)) {
            ssc[i] = c; ssc[p] = a;
            int t = sid[i]; sid[i] = sid[p]; sid[p] = t;
          }
        }
      }
      __syncthreads();
    }
  }
  for (int r = threadIdx.x; r < k; r += blockDim.x) {
    int oid = sid[r];
    newidx[(size_t)g * n + oid] = r;
    perm[(size_t)g * k + r] = g * n + oid;
    gate[(size_t)g * k + r] = ssc[r];
  }
}

__global__ void gather_gate_kernel(const float* __restrict__ xin, const int* __restrict__ perm,
                                   const float* __restrict__ gate, float* __restrict__ xout,
                                   long long cnt, int H) {
  long long idx = (long long)blockIdx.x * blockDim.x + threadIdx.x;
  if (idx >= cnt * H) return;
  long long i = idx / H;
  int f = (int)(idx - i * H);
  xout[idx] = xin[(size_t)perm[i] * H + f] * gate[i];
}

__global__ void edge_remap_kernel(const int* __restrict__ src, const int* __restrict__ dst,
                                  const float* __restrict__ ew, const int* __restrict__ newidx,
                                  int n, int k, int* __restrict__ src2, int* __restrict__ dst2,
                                  float* __restrict__ ew2, long long E) {
  long long e = (long long)blockIdx.x * blockDim.x + threadIdx.x;
  if (e >= E) return;
  int os = src[e], od = dst[e];
  int ns = newidx[os], nd = newidx[od];
  bool keep = (ns >= 0) && (nd >= 0);
  int g = os / n;
  src2[e] = keep ? g * k + ns : 0;
  dst2[e] = keep ? g * k + nd : 0;
  ew2[e]  = keep ? ew[e] : 0.0f;
}

// readout: z[g, 0:H] += max over nodes, z[g, H:2H] += mean; one block per graph
__global__ __launch_bounds__(256)
void readout_kernel(const float* __restrict__ xp, float* __restrict__ z, int k, int H) {
  __shared__ float smx[8][32];
  __shared__ float ssm[8][32];
  int g = blockIdx.x;
  int f = threadIdx.x & 31;
  int c = threadIdx.x >> 5;
  float mx = -3.4e38f, sm = 0.0f;
  if (f < H) {
    for (int r = c; r < k; r += 8) {
      float v = xp[((size_t)g * k + r) * H + f];
      mx = fmaxf(mx, v); sm += v;
    }
  }
  smx[c][f] = mx; ssm[c][f] = sm;
  __syncthreads();
  if (c == 0 && f < H) {
#pragma unroll
    for (int i = 1; i < 8; ++i) { mx = fmaxf(mx, smx[i][f]); sm += ssm[i][f]; }
    z[g * 2 * H + f]     += mx;
    z[g * 2 * H + H + f] += sm / (float)k;
  }
}

// ---------------------------- launcher ----------------------------
static inline unsigned cdiv(long long a, long long b) { return (unsigned)((a + b - 1) / b); }

extern "C" void kernel_launch(void* const* d_in, const int* in_sizes, int n_in,
                              void* d_out, int out_size, void* d_ws, size_t ws_size,
                              hipStream_t stream) {
  const float* x      = (const float*)d_in[0];
  const int*   ei     = (const int*)d_in[1];
  const float* ew0    = (const float*)d_in[2];
  const float* ln_g   = (const float*)d_in[4];
  const float* ln_b   = (const float*)d_in[5];
  const float* W_rel1 = (const float*)d_in[6];
  const float* b_rel1 = (const float*)d_in[7];
  const float* W_rt1  = (const float*)d_in[8];
  const float* W_rel2 = (const float*)d_in[9];
  const float* b_rel2 = (const float*)d_in[10];
  const float* W_rt2  = (const float*)d_in[11];
  const float* W_rel3 = (const float*)d_in[12];
  const float* b_rel3 = (const float*)d_in[13];
  const float* W_rt3  = (const float*)d_in[14];
  const float* pw1    = (const float*)d_in[15];
  const float* pw2    = (const float*)d_in[16];
  const float* pw3    = (const float*)d_in[17];
  const float* hW1    = (const float*)d_in[18];
  const float* hb1    = (const float*)d_in[19];
  const float* hW2    = (const float*)d_in[20];
  const float* hb2    = (const float*)d_in[21];
  const float* hW3    = (const float*)d_in[22];
  const float* hb3    = (const float*)d_in[23];
  const int* src0 = ei;
  const int* dst0 = ei + ETOT;

  // workspace carve-out
  unsigned char* wp = (unsigned char*)d_ws;
  auto alloc = [&](size_t bytes) -> void* {
    void* p = wp; wp += ((bytes + 255) & ~(size_t)255); return p;
  };
  float*  h     = (float*) alloc((size_t)NTOT * 128 * 4);
  float*  agg   = (float*) alloc((size_t)NTOT * 128 * 4);
  float*  xa    = (float*) alloc((size_t)NTOT * HD * 4);
  float*  xb    = (float*) alloc((size_t)NTOT * HD * 4);
  float*  sc    = (float*) alloc((size_t)NTOT * 4);
  int*    perm  = (int*)   alloc((size_t)BG * KP1 * 4);
  float*  gate  = (float*) alloc((size_t)BG * KP1 * 4);
  int*    nidx  = (int*)   alloc((size_t)NTOT * 4);
  int*    srcA  = (int*)   alloc((size_t)ETOT * 4);
  int*    dstA  = (int*)   alloc((size_t)ETOT * 4);
  float*  ewA   = (float*) alloc((size_t)ETOT * 4);
  int*    srcB  = (int*)   alloc((size_t)ETOT * 4);
  int*    dstB  = (int*)   alloc((size_t)ETOT * 4);
  float*  ewB   = (float*) alloc((size_t)ETOT * 4);
  float*  z     = (float*) alloc((size_t)BG * 2 * HD * 4);
  float*  a1    = (float*) alloc((size_t)NHD * BG * 8 * HD * 4);
  float*  a2    = (float*) alloc((size_t)NHD * BG * 32 * HD * 4);
  // bf16 staging (Kp = K rounded up to 32)
  __bf16* hb_   = (__bf16*)alloc((size_t)NTOT * 128 * 2);
  __bf16* aggb  = (__bf16*)alloc((size_t)NTOT * 128 * 2);
  __bf16* xbb   = (__bf16*)alloc((size_t)NTOT * 32 * 2);
  __bf16* zb    = (__bf16*)alloc((size_t)BG * 64 * 2);
  __bf16* a1b   = (__bf16*)alloc((size_t)NHD * BG * 256 * 2);
  __bf16* a2b   = (__bf16*)alloc((size_t)NHD * BG * 960 * 2);
  // packed weight fragments
  __bf16* Pr1 = (__bf16*)alloc((size_t)4 * 2 * 512 * 2);
  __bf16* Ps1 = (__bf16*)alloc((size_t)4 * 2 * 512 * 2);
  __bf16* Pr2 = (__bf16*)alloc((size_t)1 * 2 * 512 * 2);
  __bf16* Ps2 = (__bf16*)alloc((size_t)1 * 2 * 512 * 2);
  __bf16* Pr3 = (__bf16*)alloc((size_t)1 * 2 * 512 * 2);
  __bf16* Ps3 = (__bf16*)alloc((size_t)1 * 2 * 512 * 2);
  __bf16* pW1 = (__bf16*)alloc((size_t)NHD * 2 * 15 * 512 * 2);
  __bf16* pW2 = (__bf16*)alloc((size_t)NHD * 8 * 60 * 512 * 2);
  __bf16* pW3 = (__bf16*)alloc((size_t)NHD * 30 * 1 * 512 * 2);

  // ---- one-time weight packing (cheap, deterministic every call) ----
  pack_w_kernel<<<dim3(4, 2, 1), 32, 0, stream>>>(W_rel1, 0, Pr1, 0, 128, HD, 2);
  pack_w_kernel<<<dim3(4, 2, 1), 32, 0, stream>>>(W_rt1,  0, Ps1, 0, 128, HD, 2);
  pack_w_kernel<<<dim3(1, 2, 1), 32, 0, stream>>>(W_rel2, 0, Pr2, 0, HD, HD, 2);
  pack_w_kernel<<<dim3(1, 2, 1), 32, 0, stream>>>(W_rt2,  0, Ps2, 0, HD, HD, 2);
  pack_w_kernel<<<dim3(1, 2, 1), 32, 0, stream>>>(W_rel3, 0, Pr3, 0, HD, HD, 2);
  pack_w_kernel<<<dim3(1, 2, 1), 32, 0, stream>>>(W_rt3,  0, Ps3, 0, HD, HD, 2);
  pack_w_kernel<<<dim3(2, 15, NHD), 32, 0, stream>>>(hW1, 60 * 240, pW1, 2 * 15 * 512LL, 60, 240, 15);
  pack_w_kernel<<<dim3(8, 60, NHD), 32, 0, stream>>>(hW2, 240 * 960, pW2, 8 * 60 * 512LL, 240, 960, 60);
  pack_w_kernel<<<dim3(30, 1, NHD), 32, 0, stream>>>(hW3, 960 * 8, pW3, 30 * 512LL, 960, 8, 1);

  fill_kernel<<<cdiv(BG * 2 * HD, 256), 256, 0, stream>>>(z, 0.0f, BG * 2 * HD);

  // ---------- stage 1 (K=128) ----------
  layernorm_kernel<<<cdiv((long long)NTOT * 32, 256), 256, 0, stream>>>(x, ln_g, ln_b, h, NTOT);
  fill_kernel<<<cdiv((long long)NTOT * 128, 256), 256, 0, stream>>>(agg, 0.0f, (long long)NTOT * 128);
  scatter_kernel<<<cdiv(ETOT * 128, 256), 256, 0, stream>>>(h, src0, dst0, ew0, agg, ETOT, 128);
  cvt_pad_kernel<<<cdiv((long long)NTOT * 128, 256), 256, 0, stream>>>(h,   hb_,  NTOT, 128, 128);
  cvt_pad_kernel<<<cdiv((long long)NTOT * 128, 256), 256, 0, stream>>>(agg, aggb, NTOT, 128, 128);
  gconv_mm_kernel<<<NTOT / 16, 32, 0, stream>>>(aggb, hb_, 128, Pr1, Ps1, b_rel1, xa, HD);
  score_kernel<<<cdiv((long long)NTOT * 32, 256), 256, 0, stream>>>(xa, pw1, sc, NTOT, HD);
  topk_kernel<<<BG, 1024, 0, stream>>>(sc, NPG, KP1, nidx, perm, gate);
  gather_gate_kernel<<<cdiv((long long)BG * KP1 * HD, 256), 256, 0, stream>>>(xa, perm, gate, xb, (long long)BG * KP1, HD);
  edge_remap_kernel<<<cdiv(ETOT, 256), 256, 0, stream>>>(src0, dst0, ew0, nidx, NPG, KP1, srcA, dstA, ewA, ETOT);
  readout_kernel<<<BG, 256, 0, stream>>>(xb, z, KP1, HD);

  // ---------- stage 2 (K=30 -> Kp=32) ----------
  const int N2 = BG * KP1;
  fill_kernel<<<cdiv((long long)N2 * HD, 256), 256, 0, stream>>>(agg, 0.0f, (long long)N2 * HD);
  scatter_kernel<<<cdiv(ETOT * HD, 256), 256, 0, stream>>>(xb, srcA, dstA, ewA, agg, ETOT, HD);
  cvt_pad_kernel<<<cdiv((long long)N2 * 32, 256), 256, 0, stream>>>(xb,  xbb,  N2, HD, 32);
  cvt_pad_kernel<<<cdiv((long long)N2 * 32, 256), 256, 0, stream>>>(agg, aggb, N2, HD, 32);
  gconv_mm_kernel<<<N2 / 16, 32, 0, stream>>>(aggb, xbb, 32, Pr2, Ps2, b_rel2, xa, HD);
  score_kernel<<<cdiv((long long)N2 * 32, 256), 256, 0, stream>>>(xa, pw2, sc, N2, HD);
  topk_kernel<<<BG, 1024, 0, stream>>>(sc, KP1, KP2, nidx, perm, gate);
  gather_gate_kernel<<<cdiv((long long)BG * KP2 * HD, 256), 256, 0, stream>>>(xa, perm, gate, xb, (long long)BG * KP2, HD);
  edge_remap_kernel<<<cdiv(ETOT, 256), 256, 0, stream>>>(srcA, dstA, ewA, nidx, KP1, KP2, srcB, dstB, ewB, ETOT);
  readout_kernel<<<BG, 256, 0, stream>>>(xb, z, KP2, HD);

  // ---------- stage 3 (K=30 -> Kp=32) ----------
  const int N3 = BG * KP2;
  fill_kernel<<<cdiv((long long)N3 * HD, 256), 256, 0, stream>>>(agg, 0.0f, (long long)N3 * HD);
  scatter_kernel<<<cdiv(ETOT * HD, 256), 256, 0, stream>>>(xb, srcB, dstB, ewB, agg, ETOT, HD);
  cvt_pad_kernel<<<cdiv((long long)N3 * 32, 256), 256, 0, stream>>>(xb,  xbb,  N3, HD, 32);
  cvt_pad_kernel<<<cdiv((long long)N3 * 32, 256), 256, 0, stream>>>(agg, aggb, N3, HD, 32);
  gconv_mm_kernel<<<N3 / 16, 32, 0, stream>>>(aggb, xbb, 32, Pr3, Ps3, b_rel3, xa, HD);
  score_kernel<<<cdiv((long long)N3 * 32, 256), 256, 0, stream>>>(xa, pw3, sc, N3, HD);
  topk_kernel<<<BG, 1024, 0, stream>>>(sc, KP2, KP3, nidx, perm, gate);
  gather_gate_kernel<<<cdiv((long long)BG * KP3 * HD, 256), 256, 0, stream>>>(xa, perm, gate, xb, (long long)BG * KP3, HD);
  readout_kernel<<<BG, 256, 0, stream>>>(xb, z, KP3, HD);

  // ---------- heads ----------
  relu_kernel<<<cdiv(BG * 2 * HD, 256), 256, 0, stream>>>(z, BG * 2 * HD);
  cvt_pad_kernel<<<cdiv((long long)BG * 64, 256), 256, 0, stream>>>(z, zb, BG, 2 * HD, 64);
  // zb[64,64] @ pW1[h] -> a1[h][64,240]
  head_gemm_kernel<<<dim3(BG / 16, 15, NHD), 32, 0, stream>>>(
      zb, 0, 64, pW1, 2 * 15 * 512LL, 15, hb1, 8 * HD, a1, (long long)BG * 8 * HD, 8 * HD, 1);
  cvt_pad_kernel<<<cdiv((long long)NHD * BG * 256, 256), 256, 0, stream>>>(a1, a1b, (long long)NHD * BG, 240, 256);
  // a1b[h][64,256] @ pW2[h] -> a2[h][64,960]
  head_gemm_kernel<<<dim3(BG / 16, 60, NHD), 32, 0, stream>>>(
      a1b, (long long)BG * 256, 256, pW2, 8 * 60 * 512LL, 60, hb2, 32 * HD, a2, (long long)BG * 32 * HD, 32 * HD, 1);
  cvt_pad_kernel<<<cdiv((long long)NHD * BG * 960, 256), 256, 0, stream>>>(a2, a2b, (long long)NHD * BG, 960, 960);
  // a2b[h][64,960] @ pW3[h] -> out[h][64,8]
  head_gemm_kernel<<<dim3(BG / 16, 1, NHD), 32, 0, stream>>>(
      a2b, (long long)BG * 960, 960, pW3, 30 * 512LL, 1, hb3, ODIM, (float*)d_out, (long long)BG * ODIM, ODIM, 0);

  (void)in_sizes; (void)n_in; (void)out_size; (void)ws_size;
}
